// VQVAE_81621558493561
// MI455X (gfx1250) — compile-verified, matrix operand order checked by reading, not run on hardware
//
#include <hip/hip_runtime.h>

// ---------------- types / WMMA helpers (gfx1250, wave32) ----------------
typedef __attribute__((ext_vector_type(16))) _Float16 v16h;
typedef __attribute__((ext_vector_type(8)))  _Float16 v8h;
typedef __attribute__((ext_vector_type(8)))  float    v8f;

#define DEV __device__ __forceinline__

DEV v8f wmma32(v16h a, v16h b, v8f c) {
  // v_wmma_f32_16x16x32_f16: D = A(16x32 f16) * B(32x16 f16) + C(16x16 f32)
  return __builtin_amdgcn_wmma_f32_16x16x32_f16(false, a, false, b,
                                                (short)0, c, false, false);
}

// gfx1250 async memory->LDS copy (ASYNCcnt-tracked, bypasses VGPRs).
// LDS destination address = low 32 bits of the generic shared pointer.
DEV void async_b128(unsigned lds_off, const void* gaddr) {
  asm volatile("global_load_async_to_lds_b128 %0, %1, off"
               : : "v"(lds_off), "v"(gaddr) : "memory");
}
DEV void wait_async() {
  asm volatile("s_wait_asynccnt 0x0" : : : "memory");
}
DEV unsigned lds_addr(const void* p) { return (unsigned)(size_t)p; }

// Load a 16x32 f16 fragment from a row-major [rows][stride] LDS tile.
// 16-bit A-matrix layout (ISA 7.12.2): lanes 0-15 -> M=lane, lanes 16-31 -> M=lane-16;
// halves 0..7 = K kb..kb+7, halves 8..15 = K kb+16..kb+23, kb = 8*(lane>=16).
DEV v16h frag_rm(const _Float16* base, int row, int stride, int k0, int lane) {
  const _Float16* p = base + (size_t)(row + (lane & 15)) * stride
                            + k0 + ((lane >> 4) << 3);
  v8h lo = *reinterpret_cast<const v8h*>(p);
  v8h hi = *reinterpret_cast<const v8h*>(p + 16);
  v16h r;
#pragma unroll
  for (int i = 0; i < 8; ++i) { r[i] = lo[i]; r[i + 8] = hi[i]; }
  return r;
}

DEV int iclamp(int v, int lo, int hi) { return v < lo ? lo : (v > hi ? hi : v); }

// ---------------- problem constants ----------------
// x:(32,3,256,256)  e1->(32,32,128,128)  e2/z->(32,64,64,64)
// emb:(512,64)  d1->(32,64,128,128)  d2->(32,32,256,256)  d3->(32,3,256,256)
#define ZELEMS  8388608       // 32*64*64*64
#define RECON   6291456       // 32*3*256*256

// ================= small prep kernels =================
__global__ void k_cast_f16(const float* __restrict__ s, _Float16* __restrict__ d, int n) {
  int i = blockIdx.x * 256 + threadIdx.x;
  if (i < n) d[i] = (_Float16)s[i];
}

__global__ void k_embn2(const float* __restrict__ emb, float* __restrict__ n2) {
  int i = blockIdx.x * 256 + threadIdx.x;
  if (i < 512) {
    float s = 0.f;
#pragma unroll 8
    for (int k = 0; k < 64; ++k) { float v = emb[i * 64 + k]; s += v * v; }
    n2[i] = s;
  }
}

// pack d1_w (in=64,out=64,4,4) -> Bd1[pa][n(64)][k(256)]  k = c*4 + (a<<1|bb)
__global__ void k_pack_d1(const float* __restrict__ w, _Float16* __restrict__ Bp) {
  int i = blockIdx.x * 256 + threadIdx.x;            // 4*64*256 = 65536
  if (i >= 65536) return;
  int pa = i >> 14, n = (i >> 8) & 63, k = i & 255;
  int c = k >> 2, t = k & 3;
  int py = pa >> 1, px = pa & 1;
  int ky = py ? ((t >> 1) ? 2 : 0) : ((t >> 1) ? 3 : 1);
  int kx = px ? ((t & 1) ? 2 : 0) : ((t & 1) ? 3 : 1);
  Bp[i] = (_Float16)w[((c * 64 + n) * 4 + ky) * 4 + kx];
}

// pack d2_w (in=64,out=32,4,4) -> Bd2[pa][n(32)][k(256)]
__global__ void k_pack_d2(const float* __restrict__ w, _Float16* __restrict__ Bp) {
  int i = blockIdx.x * 256 + threadIdx.x;            // 4*32*256 = 32768
  if (i >= 32768) return;
  int pa = i >> 13, n = (i >> 8) & 31, k = i & 255;
  int c = k >> 2, t = k & 3;
  int py = pa >> 1, px = pa & 1;
  int ky = py ? ((t >> 1) ? 2 : 0) : ((t >> 1) ? 3 : 1);
  int kx = px ? ((t & 1) ? 2 : 0) : ((t & 1) ? 3 : 1);
  Bp[i] = (_Float16)w[((c * 32 + n) * 4 + ky) * 4 + kx];
}

__global__ void k_zero(float* p) { if (threadIdx.x == 0) *p = 0.f; }

// ================= e1: direct conv 3->32 k4 s2 p1, relu, f16 out ============
__global__ __launch_bounds__(256)
void k_e1(const float* __restrict__ x, const float* __restrict__ w,
          const float* __restrict__ bias, _Float16* __restrict__ h1) {
  unsigned i = blockIdx.x * 256 + threadIdx.x;       // 32*32*128*128 = 16777216
  int ox = i & 127, oy = (i >> 7) & 127, oc = (i >> 14) & 31, b = i >> 19;
  float s = bias[oc];
#pragma unroll
  for (int c = 0; c < 3; ++c)
#pragma unroll
    for (int ky = 0; ky < 4; ++ky) {
      int iy = 2 * oy + ky - 1;
      bool oky = (unsigned)iy < 256u;
      int iyc = iclamp(iy, 0, 255);
#pragma unroll
      for (int kx = 0; kx < 4; ++kx) {
        int ix = 2 * ox + kx - 1;
        bool ok = oky && (unsigned)ix < 256u;
        int ixc = iclamp(ix, 0, 255);
        float v = x[(((size_t)b * 3 + c) * 256 + iyc) * 256 + ixc];
        s += (ok ? v : 0.f) * w[((oc * 3 + c) * 4 + ky) * 4 + kx];
      }
    }
  h1[i] = (_Float16)(s > 0.f ? s : 0.f);
}

// ================= e2: implicit-GEMM WMMA, M=131072 N=64 K=512 ==============
// A[m][k] = h1[b,c,2oy+ky-1,2ox+kx-1]  (m=(b,oy,ox), k=(c,ky,kx))
// B packed n-major: Bw[n*512 + k] (== e2_w flat, cast to f16)
__global__ __launch_bounds__(256)
void k_e2(const _Float16* __restrict__ h1, const _Float16* __restrict__ Bw,
          const float* __restrict__ bias,
          float* __restrict__ z_nchw, _Float16* __restrict__ z_rm) {
  __shared__ __align__(16) _Float16 As[128][32];
  __shared__ __align__(16) _Float16 Bs[64][32];
  const int tid = threadIdx.x, lane = tid & 31, wave = tid >> 5;
  const int wm = wave >> 1, wn = wave & 1;           // 4x2 waves, 32x32 each
  const int m0 = blockIdx.x * 128;

  v8f acc[2][2] = {};
  const int arow = tid >> 1, ahb = (tid & 1) * 16;
  const int am = m0 + arow;
  const int ab = am >> 12, arem = am & 4095;
  const int aoy = arem >> 6, aox = arem & 63;
  const int bn = tid >> 2, bhb = (tid & 3) * 8;
  const unsigned ldsB = lds_addr(&Bs[bn][bhb]);

  for (int kc = 0; kc < 16; ++kc) {
    const int ks0 = kc * 32;
    // stage B: async memory->LDS (ASYNCcnt path), fully coalesced n-major
    {
      const _Float16* src = Bw + (size_t)bn * 512 + ks0 + bhb;
      async_b128(ldsB, src);
      __builtin_prefetch((const void*)(src + 32), 0, 1);
    }
    // stage A (im2col gather, branch-free clamped loads)
#pragma unroll
    for (int i = 0; i < 16; ++i) {
      int k = ks0 + ahb + i;
      int c = k >> 4, ky = (k >> 2) & 3, kx = k & 3;
      int iy = 2 * aoy + ky - 1, ix = 2 * aox + kx - 1;
      bool ok = (unsigned)iy < 128u && (unsigned)ix < 128u;
      int iyc = iclamp(iy, 0, 127), ixc = iclamp(ix, 0, 127);
      _Float16 v = h1[(((size_t)ab * 32 + c) * 128 + iyc) * 128 + ixc];
      As[arow][ahb + i] = ok ? v : (_Float16)0.f;
    }
    wait_async();
    __syncthreads();
    v16h bf[2];
#pragma unroll
    for (int ni = 0; ni < 2; ++ni)
      bf[ni] = frag_rm(&Bs[0][0], wn * 32 + ni * 16, 32, 0, lane);
#pragma unroll
    for (int mi = 0; mi < 2; ++mi) {
      v16h af = frag_rm(&As[0][0], wm * 32 + mi * 16, 32, 0, lane);
#pragma unroll
      for (int ni = 0; ni < 2; ++ni)
        acc[mi][ni] = wmma32(af, bf[ni], acc[mi][ni]);
    }
    __syncthreads();
  }
  // epilogue: bias+relu -> z (NCHW f32) and z row-major f16
  const int hs = lane >> 4, ncol = lane & 15;
#pragma unroll
  for (int mi = 0; mi < 2; ++mi)
#pragma unroll
    for (int ni = 0; ni < 2; ++ni) {
      int n = wn * 32 + ni * 16 + ncol;
      float bv = bias[n];
#pragma unroll
      for (int r = 0; r < 8; ++r) {
        int m = m0 + wm * 32 + mi * 16 + r + 8 * hs;
        float v = acc[mi][ni][r] + bv;
        v = v > 0.f ? v : 0.f;
        int b = m >> 12, rem = m & 4095;
        z_nchw[((size_t)b * 64 + n) * 4096 + rem] = v;
        z_rm[(size_t)m * 64 + n] = (_Float16)v;
      }
    }
}

// ================= VQ: WMMA scores + wave32 shfl argmin =====================
// score[m][n] = ||emb_n||^2 - 2 * dot(z_m, emb_n); argmin over 512 codes
__global__ __launch_bounds__(256)
void k_vq(const _Float16* __restrict__ z_rm, const _Float16* __restrict__ embT,
          const float* __restrict__ embn2, int* __restrict__ idx) {
  __shared__ __align__(16) _Float16 As[256][64];   // 32 KB
  __shared__ __align__(16) _Float16 Bs[128][64];   // 16 KB, 4 code groups
  const int tid = threadIdx.x, lane = tid & 31, wave = tid >> 5;
  const int m0 = blockIdx.x * 256;

  // async-stage the whole A tile (256x64 f16 = 32 KB)
  {
    const unsigned ldsA = lds_addr(&As[0][0]);
    const char* gA = (const char*)(z_rm + (size_t)m0 * 64);
    for (int i = tid; i < 2048; i += 256)
      async_b128(ldsA + (unsigned)i * 16u, gA + (size_t)i * 16);
    wait_async();
  }
  __syncthreads();

  v16h a[2][2];
#pragma unroll
  for (int mi = 0; mi < 2; ++mi)
#pragma unroll
    for (int ks = 0; ks < 2; ++ks)
      a[mi][ks] = frag_rm(&As[0][0], wave * 32 + mi * 16, 64, ks * 32, lane);

  float bestV[16];
  int   bestI[16];
#pragma unroll
  for (int s = 0; s < 16; ++s) { bestV[s] = 3.4e38f; bestI[s] = 0; }

  const unsigned ldsBb = lds_addr(&Bs[0][0]);
  for (int g = 0; g < 4; ++g) {                    // 128 codes per group
    __syncthreads();
    {
      const char* gB = (const char*)(embT + (size_t)g * 128 * 64);
      for (int i = tid; i < 1024; i += 256)
        async_b128(ldsBb + (unsigned)i * 16u, gB + (size_t)i * 16);
      wait_async();
    }
    __syncthreads();
#pragma unroll
    for (int nc = 0; nc < 4; ++nc) {               // 32 cols per chunk
#pragma unroll
      for (int ni = 0; ni < 2; ++ni) {
        v16h b0 = frag_rm(&Bs[0][0], nc * 32 + ni * 16, 64, 0,  lane);
        v16h b1 = frag_rm(&Bs[0][0], nc * 32 + ni * 16, 64, 32, lane);
        int n = g * 128 + nc * 32 + ni * 16 + (lane & 15);
        float c2 = embn2[n];
#pragma unroll
        for (int mi = 0; mi < 2; ++mi) {
          v8f acc = {};
          acc = wmma32(a[mi][0], b0, acc);
          acc = wmma32(a[mi][1], b1, acc);
#pragma unroll
          for (int r = 0; r < 8; ++r) {
            float sc = c2 - 2.f * acc[r];
            int slot = mi * 8 + r;
            if (sc < bestV[slot]) { bestV[slot] = sc; bestI[slot] = n; }
          }
        }
      }
    }
  }
  // reduce across the 16 lanes that share each row (xor 1,2,4,8 keeps lane>>4)
#pragma unroll
  for (int s = 0; s < 16; ++s) {
    float v = bestV[s]; int bi = bestI[s];
#pragma unroll
    for (int off = 1; off < 16; off <<= 1) {
      float ov = __shfl_xor(v, off, 32);
      int   oi = __shfl_xor(bi, off, 32);
      if (ov < v || (ov == v && oi < bi)) { v = ov; bi = oi; }
    }
    if ((lane & 15) == 0) {
      int row = wave * 32 + (s >> 3) * 16 + (s & 7) + 8 * (lane >> 4);
      idx[m0 + row] = bi;
    }
  }
}

// ======== gather codes -> q (f32 out, f16 for decoder) + loss partial =======
__global__ __launch_bounds__(256)
void k_gather(const int* __restrict__ idx, const float* __restrict__ emb,
              const float* __restrict__ z_nchw, float* __restrict__ q_out,
              _Float16* __restrict__ q_f16, float* __restrict__ lossAcc) {
  __shared__ float red[256];
  size_t i = (size_t)blockIdx.x * 256 + threadIdx.x;   // ZELEMS exact
  int m = (int)(i >> 6), c = (int)(i & 63);
  float qv = emb[(size_t)idx[m] * 64 + c];
  q_out[i] = qv;
  q_f16[i] = (_Float16)qv;
  float d = qv - z_nchw[i];
  red[threadIdx.x] = d * d;
  __syncthreads();
  for (int s = 128; s > 0; s >>= 1) {
    if (threadIdx.x < (unsigned)s) red[threadIdx.x] += red[threadIdx.x + s];
    __syncthreads();
  }
  if (threadIdx.x == 0) atomicAdd(lossAcc, red[0]);
}

__global__ void k_loss(const float* acc, float* out) {
  if (threadIdx.x == 0) *out = 1.25f * (*acc) / (float)ZELEMS;
}

// ===== d1: convT 64->64 k4 s2 p1 as per-parity implicit GEMM (K=256,N=64) ===
__global__ __launch_bounds__(256)
void k_d1(const _Float16* __restrict__ q, const _Float16* __restrict__ Bw,
          const float* __restrict__ bias, _Float16* __restrict__ out) {
  const int pa = blockIdx.y, py = pa >> 1, px = pa & 1;
  const int dyt[2] = { py ? 1 : 0, py ? 0 : -1 };
  const int dxt[2] = { px ? 1 : 0, px ? 0 : -1 };
  __shared__ __align__(16) _Float16 As[128][32];
  __shared__ __align__(16) _Float16 Bs[64][32];
  const int tid = threadIdx.x, lane = tid & 31, wave = tid >> 5;
  const int wm = wave >> 1, wn = wave & 1;
  const int m0 = blockIdx.x * 128;
  const _Float16* Bp = Bw + (size_t)pa * 64 * 256;

  v8f acc[2][2] = {};
  const int arow = tid >> 1, ahb = (tid & 1) * 16;
  const int am = m0 + arow;
  const int ab = am >> 12, arem = am & 4095;
  const int au = arem >> 6, av = arem & 63;
  const int bn = tid >> 2, bhb = (tid & 3) * 8;
  const unsigned ldsB = lds_addr(&Bs[bn][bhb]);

  for (int kc = 0; kc < 8; ++kc) {
    const int ks0 = kc * 32;
    async_b128(ldsB, Bp + (size_t)bn * 256 + ks0 + bhb);
#pragma unroll
    for (int i = 0; i < 16; ++i) {
      int k = ks0 + ahb + i;
      int c = k >> 2, t = k & 3;
      int iy = au + dyt[t >> 1], ix = av + dxt[t & 1];
      bool ok = (unsigned)iy < 64u && (unsigned)ix < 64u;
      int iyc = iclamp(iy, 0, 63), ixc = iclamp(ix, 0, 63);
      _Float16 val = q[(((size_t)ab * 64 + c) * 64 + iyc) * 64 + ixc];
      As[arow][ahb + i] = ok ? val : (_Float16)0.f;
    }
    wait_async();
    __syncthreads();
    v16h bf[2];
#pragma unroll
    for (int ni = 0; ni < 2; ++ni)
      bf[ni] = frag_rm(&Bs[0][0], wn * 32 + ni * 16, 32, 0, lane);
#pragma unroll
    for (int mi = 0; mi < 2; ++mi) {
      v16h af = frag_rm(&As[0][0], wm * 32 + mi * 16, 32, 0, lane);
#pragma unroll
      for (int ni = 0; ni < 2; ++ni)
        acc[mi][ni] = wmma32(af, bf[ni], acc[mi][ni]);
    }
    __syncthreads();
  }
  const int hs = lane >> 4, ncol = lane & 15;
#pragma unroll
  for (int mi = 0; mi < 2; ++mi)
#pragma unroll
    for (int ni = 0; ni < 2; ++ni) {
      int n = wn * 32 + ni * 16 + ncol;
      float bv = bias[n];
#pragma unroll
      for (int r = 0; r < 8; ++r) {
        int m = m0 + wm * 32 + mi * 16 + r + 8 * hs;
        int b = m >> 12, rem = m & 4095;
        int u = rem >> 6, vv = rem & 63;
        float v = acc[mi][ni][r] + bv;
        v = v > 0.f ? v : 0.f;
        out[(((size_t)b * 64 + n) * 128 + 2 * u + py) * 128 + 2 * vv + px] =
            (_Float16)v;
      }
    }
}

// ===== d2: convT 64->32 k4 s2 p1, per-parity GEMM (M=524288,N=32,K=256) =====
__global__ __launch_bounds__(256)
void k_d2(const _Float16* __restrict__ h, const _Float16* __restrict__ Bw,
          const float* __restrict__ bias, _Float16* __restrict__ out) {
  const int pa = blockIdx.y, py = pa >> 1, px = pa & 1;
  const int dyt[2] = { py ? 1 : 0, py ? 0 : -1 };
  const int dxt[2] = { px ? 1 : 0, px ? 0 : -1 };
  __shared__ __align__(16) _Float16 As[256][32];
  __shared__ __align__(16) _Float16 Bs[32][32];
  const int tid = threadIdx.x, lane = tid & 31, wave = tid >> 5; // 8 waves on M
  const int m0 = blockIdx.x * 256;
  const _Float16* Bp = Bw + (size_t)pa * 32 * 256;

  v8f acc[2][2] = {};
  const int am = m0 + tid;
  const int ab = am >> 14, arem = am & 16383;
  const int au = arem >> 7, av = arem & 127;
  const int bn = tid >> 2, bhb = (tid & 3) * 8;      // valid for tid < 128
  const unsigned ldsB = lds_addr(&Bs[bn & 31][bhb]);

  for (int kc = 0; kc < 8; ++kc) {
    const int ks0 = kc * 32;
    if (tid < 128)
      async_b128(ldsB, Bp + (size_t)bn * 256 + ks0 + bhb);
#pragma unroll 4
    for (int i = 0; i < 32; ++i) {
      int k = ks0 + i;
      int c = k >> 2, t = k & 3;
      int iy = au + dyt[t >> 1], ix = av + dxt[t & 1];
      bool ok = (unsigned)iy < 128u && (unsigned)ix < 128u;
      int iyc = iclamp(iy, 0, 127), ixc = iclamp(ix, 0, 127);
      _Float16 val = h[(((size_t)ab * 64 + c) * 128 + iyc) * 128 + ixc];
      As[tid][i] = ok ? val : (_Float16)0.f;
    }
    wait_async();
    __syncthreads();
    v16h bf[2];
#pragma unroll
    for (int ni = 0; ni < 2; ++ni)
      bf[ni] = frag_rm(&Bs[0][0], ni * 16, 32, 0, lane);
#pragma unroll
    for (int mi = 0; mi < 2; ++mi) {
      v16h af = frag_rm(&As[0][0], wave * 32 + mi * 16, 32, 0, lane);
#pragma unroll
      for (int ni = 0; ni < 2; ++ni)
        acc[mi][ni] = wmma32(af, bf[ni], acc[mi][ni]);
    }
    __syncthreads();
  }
  const int hs = lane >> 4, ncol = lane & 15;
#pragma unroll
  for (int mi = 0; mi < 2; ++mi)
#pragma unroll
    for (int ni = 0; ni < 2; ++ni) {
      int n = ni * 16 + ncol;
      float bv = bias[n];
#pragma unroll
      for (int r = 0; r < 8; ++r) {
        int m = m0 + wave * 32 + mi * 16 + r + 8 * hs;
        int b = m >> 14, rem = m & 16383;
        int u = rem >> 7, vv = rem & 127;
        float v = acc[mi][ni][r] + bv;
        v = v > 0.f ? v : 0.f;
        out[(((size_t)b * 32 + n) * 256 + 2 * u + py) * 256 + 2 * vv + px] =
            (_Float16)v;
      }
    }
}

// ================= d3: direct conv 32->3 k3 p1 + sigmoid ====================
__global__ __launch_bounds__(256)
void k_d3(const _Float16* __restrict__ h, const float* __restrict__ w,
          const float* __restrict__ bias, float* __restrict__ out) {
  __shared__ float ws[3 * 32 * 9];
  for (int i = threadIdx.x; i < 864; i += 256) ws[i] = w[i];
  __syncthreads();
  int sp = blockIdx.x * 256 + threadIdx.x;   // 0..65535 spatial
  int bc = blockIdx.y;                       // 0..95 = b*3 + oc
  int b = bc / 3, oc = bc % 3;
  int ox = sp & 255, oy = sp >> 8;
  float s = bias[oc];
  for (int c = 0; c < 32; ++c)
#pragma unroll
    for (int ky = 0; ky < 3; ++ky) {
      int iy = oy + ky - 1;
      bool oky = (unsigned)iy < 256u;
      int iyc = iclamp(iy, 0, 255);
#pragma unroll
      for (int kx = 0; kx < 3; ++kx) {
        int ix = ox + kx - 1;
        bool ok = oky && (unsigned)ix < 256u;
        int ixc = iclamp(ix, 0, 255);
        float v = (float)h[(((size_t)b * 32 + c) * 256 + iyc) * 256 + ixc];
        s += (ok ? v : 0.f) * ws[(oc * 32 + c) * 9 + ky * 3 + kx];
      }
    }
  out[(size_t)bc * 65536 + sp] = 1.f / (1.f + __expf(-s));
}

// ================================ host ======================================
extern "C" void kernel_launch(void* const* d_in, const int* in_sizes, int n_in,
                              void* d_out, int out_size, void* d_ws, size_t ws_size,
                              hipStream_t stream) {
  (void)in_sizes; (void)n_in; (void)out_size; (void)ws_size;
  const float* x    = (const float*)d_in[0];
  const float* e1w  = (const float*)d_in[1];
  const float* e1b  = (const float*)d_in[2];
  const float* e2w  = (const float*)d_in[3];
  const float* e2b  = (const float*)d_in[4];
  const float* emb  = (const float*)d_in[5];
  const float* d1w  = (const float*)d_in[6];
  const float* d1b  = (const float*)d_in[7];
  const float* d2w  = (const float*)d_in[8];
  const float* d2b  = (const float*)d_in[9];
  const float* d3w  = (const float*)d_in[10];
  const float* d3b  = (const float*)d_in[11];

  float* recon = (float*)d_out;                 // 6291456
  float* lossp = recon + RECON;                 // 1
  float* q_out = lossp + 1;                     // 8388608 (q_st forward value)

  char* ws = (char*)d_ws;
  _Float16* h1    = (_Float16*)(ws);                         // 32 MB
  float*    z     = (float*)   (ws + 33554432);              // 32 MB
  _Float16* z_rm  = (_Float16*)(ws + 67108864);              // 16 MB
  _Float16* q16   = (_Float16*)(ws + 83886080);              // 16 MB
  _Float16* hd1   = (_Float16*)(ws + 100663296);             // 64 MB
  _Float16* hd2   = (_Float16*)(ws + 167772160);             // 128 MB
  _Float16* bw2   = (_Float16*)(ws + 301989888);             // 64 KB
  _Float16* embT  = (_Float16*)(ws + 302055424);             // 64 KB
  float*    embn2 = (float*)   (ws + 302120960);             // 2 KB
  _Float16* bd1   = (_Float16*)(ws + 302123008);             // 128 KB
  _Float16* bd2   = (_Float16*)(ws + 302254080);             // 64 KB
  int*      idx   = (int*)     (ws + 302319616);             // 512 KB
  float*    lacc  = (float*)   (ws + 302843904);             // 4 B

  // weight packing (tiny)
  k_cast_f16<<<(64 * 512 + 255) / 256, 256, 0, stream>>>(e2w, bw2, 64 * 512);
  k_cast_f16<<<(512 * 64 + 255) / 256, 256, 0, stream>>>(emb, embT, 512 * 64);
  k_embn2  <<<2, 256, 0, stream>>>(emb, embn2);
  k_pack_d1<<<256, 256, 0, stream>>>(d1w, bd1);
  k_pack_d2<<<128, 256, 0, stream>>>(d2w, bd2);
  k_zero   <<<1, 64, 0, stream>>>(lacc);

  // encoder
  k_e1<<<65536, 256, 0, stream>>>(x, e1w, e1b, h1);
  k_e2<<<1024, 256, 0, stream>>>(h1, bw2, e2b, z, z_rm);

  // vector quantization
  k_vq<<<512, 256, 0, stream>>>(z_rm, embT, embn2, idx);
  k_gather<<<ZELEMS / 256, 256, 0, stream>>>(idx, emb, z, q_out, q16, lacc);
  k_loss<<<1, 64, 0, stream>>>(lacc, lossp);

  // decoder
  k_d1<<<dim3(1024, 4), 256, 0, stream>>>(q16, bd1, d1b, hd1);
  k_d2<<<dim3(2048, 4), 256, 0, stream>>>(hd1, bd2, d2b, hd2);
  k_d3<<<dim3(256, 96), 256, 0, stream>>>(hd2, d3w, d3b, recon);
}
